// MultiheadAttention_85117661872173
// MI455X (gfx1250) — matrix operand order repaired and checked by reading, not compiled, not run on hardware
//
#include <hip/hip_runtime.h>

#define EMB   768
#define NH    12
#define HD    64
#define TSEQ  1024
#define BB    8
#define MTOT  (BB * TSEQ)   // 8192
#define N_QKV (3 * EMB)     // 2304

typedef __bf16 v16bf __attribute__((ext_vector_type(16)));
typedef float  v8f   __attribute__((ext_vector_type(8)));
typedef unsigned int u32x4 __attribute__((ext_vector_type(4)));

// Load a 16-element bf16 fragment as two contiguous 16-byte chunks.
__device__ __forceinline__ v16bf load_frag16(const __bf16* p0, const __bf16* p1) {
    union { v16bf v; u32x4 q[2]; } u;
    u.q[0] = *(const u32x4*)p0;
    u.q[1] = *(const u32x4*)p1;
    return u.v;
}

__device__ __forceinline__ v8f wmma_bf16(v16bf a, v16bf b, v8f c) {
    return __builtin_amdgcn_wmma_f32_16x16x32_bf16(
        /*neg_a=*/false, a, /*neg_b=*/false, b,
        /*c_mod=*/(short)0, c, /*reuse_a=*/false, /*reuse_b=*/false);
}

// ---------------- prepass: fp32 -> bf16 conversions ----------------

__global__ void convert_x_kernel(const float* __restrict__ x,
                                 __bf16* __restrict__ out, int n) {
    int i = blockIdx.x * blockDim.x + threadIdx.x;
    if (i < n) out[i] = (__bf16)x[i];
}

// w: [K,N] row-major fp32 -> wt: [N,K] bf16
__global__ void transpose_w_kernel(const float* __restrict__ w,
                                   __bf16* __restrict__ wt, int K, int N) {
    int i = blockIdx.x * blockDim.x + threadIdx.x;
    if (i < K * N) {
        int n = i / K, k = i % K;
        wt[(size_t)n * K + k] = (__bf16)w[(size_t)k * N + n];
    }
}

// ---------------- QKV projection GEMM ----------------
// C[8192,2304] = X[8192,768] @ W + b ; scatter into Q,K [B,H,T,D] and Vt [B,H,D,T]
// block = 128 threads (4 waves); wave -> 32x64 tile (2 A frags x 4 B frags, 8 WMMA/step),
// register double-buffered K loop. grid (256, 9)

__global__ __launch_bounds__(128) void qkv_gemm_kernel(
    const __bf16* __restrict__ X, const __bf16* __restrict__ Wt,
    const float* __restrict__ bias,
    __bf16* __restrict__ Q, __bf16* __restrict__ Km, __bf16* __restrict__ Vt)
{
    const int lane = threadIdx.x & 31;
    const int wave = threadIdx.x >> 5;
    const int rowBase = blockIdx.x * 32;
    const int colBase = blockIdx.y * 256 + wave * 64;
    const int l16 = lane & 15;
    const int hi  = lane >> 4;

    const __bf16* xrow0 = X + (size_t)(rowBase + l16) * EMB;
    const __bf16* xrow1 = xrow0 + (size_t)16 * EMB;

    auto loadA = [&](const __bf16* p, int k0) {
        return load_frag16(p + k0 + 8 * hi, p + k0 + 16 + 8 * hi);
    };
    auto loadB = [&](int t, int k0) {
        const __bf16* w = Wt + (size_t)(colBase + 16 * t + l16) * EMB + k0 + 8 * hi;
        return load_frag16(w, w + 16);
    };

    v8f acc0[4] = {}, acc1[4] = {};

    v16bf a0 = loadA(xrow0, 0), a1 = loadA(xrow1, 0);
    v16bf b0 = loadB(0, 0), b1 = loadB(1, 0), b2 = loadB(2, 0), b3 = loadB(3, 0);

    for (int k0 = 0; k0 < EMB; k0 += 32) {
        const int kn = k0 + 32;
        v16bf na0, na1, nb0, nb1, nb2, nb3;
        if (kn < EMB) {                    // issue next-step loads first
            na0 = loadA(xrow0, kn); na1 = loadA(xrow1, kn);
            nb0 = loadB(0, kn); nb1 = loadB(1, kn);
            nb2 = loadB(2, kn); nb3 = loadB(3, kn);
        }
        acc0[0] = wmma_bf16(a0, b0, acc0[0]);
        acc0[1] = wmma_bf16(a0, b1, acc0[1]);
        acc0[2] = wmma_bf16(a0, b2, acc0[2]);
        acc0[3] = wmma_bf16(a0, b3, acc0[3]);
        acc1[0] = wmma_bf16(a1, b0, acc1[0]);
        acc1[1] = wmma_bf16(a1, b1, acc1[1]);
        acc1[2] = wmma_bf16(a1, b2, acc1[2]);
        acc1[3] = wmma_bf16(a1, b3, acc1[3]);
        if (kn < EMB) {
            a0 = na0; a1 = na1;
            b0 = nb0; b1 = nb1; b2 = nb2; b3 = nb3;
        }
    }

#pragma unroll
    for (int t = 0; t < 4; ++t) {
        int col   = colBase + 16 * t + l16;
        int which = col / EMB;       // 0=Q 1=K 2=V (uniform per 16-wide tile)
        int c     = col % EMB;
        int h = c / HD, d = c % HD;
        float bv = bias[col];
#pragma unroll
        for (int half = 0; half < 2; ++half) {
            const v8f& acc = half ? acc1[t] : acc0[t];
#pragma unroll
            for (int r = 0; r < 8; ++r) {
                int row = rowBase + 16 * half + r + 8 * hi;
                int b_i = row >> 10, tt = row & (TSEQ - 1);
                size_t bh = (size_t)(b_i * NH + h);
                __bf16 val = (__bf16)(acc[r] + bv);
                if (which == 0)      Q [(bh * TSEQ + tt) * HD + d] = val;
                else if (which == 1) Km[(bh * TSEQ + tt) * HD + d] = val;
                else                 Vt[(bh * HD + d) * TSEQ + tt] = val;
            }
        }
    }
}

// ---------------- fused causal flash attention ----------------
// one wave (32 threads) per (b,h, 16-row q tile); key tiles of 32;
// next-tile K/V prefetched via global_prefetch

__global__ __launch_bounds__(32) void flash_attn_kernel(
    const __bf16* __restrict__ Q, const __bf16* __restrict__ Km,
    const __bf16* __restrict__ Vt, __bf16* __restrict__ O)
{
    __shared__ __bf16 lP[16 * 32];   // P tile staging, C-layout -> A-layout
    const int lane = threadIdx.x & 31;
    const int l16 = lane & 15, hi = lane >> 4;
    const int bh    = blockIdx.x >> 6;
    const int qBase = (blockIdx.x & 63) * 16;
    const int b_i = bh / NH, h = bh % NH;

    const __bf16* Qb = Q  + (size_t)bh * TSEQ * HD;
    const __bf16* Kb = Km + (size_t)bh * TSEQ * HD;
    const __bf16* Vb = Vt + (size_t)bh * HD * TSEQ;

    // Q: two 16x32 A fragments (d = 0..31, 32..63)
    v16bf qa[2];
    {
        const __bf16* qrow = Qb + (size_t)(qBase + l16) * HD;
#pragma unroll
        for (int c2 = 0; c2 < 2; ++c2)
            qa[c2] = load_frag16(qrow + 32 * c2 + 8 * hi,
                                 qrow + 32 * c2 + 16 + 8 * hi);
    }

    float m[8], l[8];
    v8f o[4] = {};
#pragma unroll
    for (int r = 0; r < 8; ++r) { m[r] = -1e30f; l[r] = 0.f; }

    const float scale = 0.125f;                 // 1/sqrt(64)
    const int nK = (qBase + 16 + 31) >> 5;      // causal: keys <= qBase+15

    for (int kt = 0; kt < nK; ++kt) {
        const int kBase = kt * 32;

        // prefetch next key tile (K rows + V rows) one iteration ahead
        if (kt + 1 < nK) {
            const int nb = kBase + 32;
            __builtin_prefetch(Kb + (size_t)(nb + lane) * HD, 0, 0);
            __builtin_prefetch(Vb + (size_t)lane * TSEQ + nb, 0, 0);
            __builtin_prefetch(Vb + (size_t)(32 + lane) * TSEQ + nb, 0, 0);
        }

        // S = Q @ K^T : 16x32 tile (two 16x16 accumulators)
        v8f s[2] = {};
#pragma unroll
        for (int j = 0; j < 2; ++j) {
#pragma unroll
            for (int c2 = 0; c2 < 2; ++c2) {
                const __bf16* krow =
                    Kb + (size_t)(kBase + 16 * j + l16) * HD + 32 * c2 + 8 * hi;
                v16bf kf = load_frag16(krow, krow + 16);
                s[j] = wmma_bf16(qa[c2], kf, s[j]);
            }
        }

        // scale + causal mask
        float p[2][8];
#pragma unroll
        for (int j = 0; j < 2; ++j) {
            int key = kBase + 16 * j + l16;
#pragma unroll
            for (int r = 0; r < 8; ++r) {
                int qrow = qBase + r + 8 * hi;
                float xv = s[j][r] * scale;
                p[j][r] = (key <= qrow) ? xv : -1e30f;
            }
        }

        __syncthreads();   // lP reuse guard
        // online softmax per row (rows live in 16-lane halves matching C layout)
#pragma unroll
        for (int r = 0; r < 8; ++r) {
            float tmax = fmaxf(p[0][r], p[1][r]);
#pragma unroll
            for (int mk = 1; mk < 16; mk <<= 1)
                tmax = fmaxf(tmax, __shfl_xor(tmax, mk, 32));
            float mnew  = fmaxf(m[r], tmax);
            float alpha = __expf(m[r] - mnew);
            float e0 = __expf(p[0][r] - mnew);
            float e1 = __expf(p[1][r] - mnew);
            float rsum = e0 + e1;
#pragma unroll
            for (int mk = 1; mk < 16; mk <<= 1)
                rsum += __shfl_xor(rsum, mk, 32);
            l[r] = l[r] * alpha + rsum;
            m[r] = mnew;
#pragma unroll
            for (int t = 0; t < 4; ++t) o[t][r] *= alpha;
            // store P in C layout: row r+8*hi, cols l16 / l16+16
            int row = r + 8 * hi;
            lP[row * 32 + l16]      = (__bf16)e0;
            lP[row * 32 + 16 + l16] = (__bf16)e1;
        }
        __syncthreads();

        // reload P as 16x32 A fragment, then O += P @ V
        v16bf pa = load_frag16(&lP[l16 * 32 + 8 * hi],
                               &lP[l16 * 32 + 16 + 8 * hi]);
#pragma unroll
        for (int t = 0; t < 4; ++t) {
            const __bf16* vrow =
                Vb + (size_t)(16 * t + l16) * TSEQ + kBase + 8 * hi;
            v16bf vf = load_frag16(vrow, vrow + 16);
            o[t] = wmma_bf16(pa, vf, o[t]);
        }
    }

    // normalize + write [B,T,C] bf16
#pragma unroll
    for (int t = 0; t < 4; ++t) {
        int d = 16 * t + l16;
#pragma unroll
        for (int r = 0; r < 8; ++r) {
            int trow = qBase + r + 8 * hi;
            float val = o[t][r] / l[r];
            O[((size_t)b_i * TSEQ + trow) * EMB + h * HD + d] = (__bf16)val;
        }
    }
}

// ---------------- output projection GEMM ----------------
// out[8192,768] fp32 = A[8192,768]bf16 @ WtP + b ; wave -> 32x64 tile,
// register double-buffered. grid (256, 3), 128 threads

__global__ __launch_bounds__(128) void proj_gemm_kernel(
    const __bf16* __restrict__ A, const __bf16* __restrict__ Wt,
    const float* __restrict__ bias, float* __restrict__ out)
{
    const int lane = threadIdx.x & 31;
    const int wave = threadIdx.x >> 5;
    const int rowBase = blockIdx.x * 32;
    const int colBase = blockIdx.y * 256 + wave * 64;
    const int l16 = lane & 15, hi = lane >> 4;

    const __bf16* arow0 = A + (size_t)(rowBase + l16) * EMB;
    const __bf16* arow1 = arow0 + (size_t)16 * EMB;

    auto loadA = [&](const __bf16* p, int k0) {
        return load_frag16(p + k0 + 8 * hi, p + k0 + 16 + 8 * hi);
    };
    auto loadB = [&](int t, int k0) {
        const __bf16* w = Wt + (size_t)(colBase + 16 * t + l16) * EMB + k0 + 8 * hi;
        return load_frag16(w, w + 16);
    };

    v8f acc0[4] = {}, acc1[4] = {};
    v16bf a0 = loadA(arow0, 0), a1 = loadA(arow1, 0);
    v16bf b0 = loadB(0, 0), b1 = loadB(1, 0), b2 = loadB(2, 0), b3 = loadB(3, 0);

    for (int k0 = 0; k0 < EMB; k0 += 32) {
        const int kn = k0 + 32;
        v16bf na0, na1, nb0, nb1, nb2, nb3;
        if (kn < EMB) {
            na0 = loadA(arow0, kn); na1 = loadA(arow1, kn);
            nb0 = loadB(0, kn); nb1 = loadB(1, kn);
            nb2 = loadB(2, kn); nb3 = loadB(3, kn);
        }
        acc0[0] = wmma_bf16(a0, b0, acc0[0]);
        acc0[1] = wmma_bf16(a0, b1, acc0[1]);
        acc0[2] = wmma_bf16(a0, b2, acc0[2]);
        acc0[3] = wmma_bf16(a0, b3, acc0[3]);
        acc1[0] = wmma_bf16(a1, b0, acc1[0]);
        acc1[1] = wmma_bf16(a1, b1, acc1[1]);
        acc1[2] = wmma_bf16(a1, b2, acc1[2]);
        acc1[3] = wmma_bf16(a1, b3, acc1[3]);
        if (kn < EMB) {
            a0 = na0; a1 = na1;
            b0 = nb0; b1 = nb1; b2 = nb2; b3 = nb3;
        }
    }

#pragma unroll
    for (int t = 0; t < 4; ++t) {
        int col = colBase + 16 * t + l16;
        float bv = bias[col];
#pragma unroll
        for (int half = 0; half < 2; ++half) {
            const v8f& acc = half ? acc1[t] : acc0[t];
#pragma unroll
            for (int r = 0; r < 8; ++r) {
                int row = rowBase + 16 * half + r + 8 * hi;
                out[(size_t)row * EMB + col] = acc[r] + bv;
            }
        }
    }
}

// ---------------- launcher ----------------

extern "C" void kernel_launch(void* const* d_in, const int* in_sizes, int n_in,
                              void* d_out, int out_size, void* d_ws, size_t ws_size,
                              hipStream_t stream) {
    const float* x      = (const float*)d_in[0];
    const float* w_attn = (const float*)d_in[1];
    const float* b_attn = (const float*)d_in[2];
    const float* w_proj = (const float*)d_in[3];
    const float* b_proj = (const float*)d_in[4];

    char* ws = (char*)d_ws;
    size_t off = 0;
    auto alloc = [&](size_t bytes) -> void* {
        void* p = ws + off;
        off += (bytes + 255) & ~(size_t)255;
        return p;
    };
    __bf16* X16 = (__bf16*)alloc((size_t)MTOT * EMB * 2);
    __bf16* WtA = (__bf16*)alloc((size_t)N_QKV * EMB * 2);
    __bf16* WtP = (__bf16*)alloc((size_t)EMB * EMB * 2);
    __bf16* Qd  = (__bf16*)alloc((size_t)MTOT * EMB * 2);
    __bf16* Kd  = (__bf16*)alloc((size_t)MTOT * EMB * 2);
    __bf16* Vtd = (__bf16*)alloc((size_t)MTOT * EMB * 2);
    __bf16* AO  = (__bf16*)alloc((size_t)MTOT * EMB * 2);

    {
        int n = MTOT * EMB;
        convert_x_kernel<<<(n + 255) / 256, 256, 0, stream>>>(x, X16, n);
    }
    {
        int n = EMB * N_QKV;
        transpose_w_kernel<<<(n + 255) / 256, 256, 0, stream>>>(w_attn, WtA, EMB, N_QKV);
    }
    {
        int n = EMB * EMB;
        transpose_w_kernel<<<(n + 255) / 256, 256, 0, stream>>>(w_proj, WtP, EMB, EMB);
    }

    dim3 g1(MTOT / 32, N_QKV / 256);   // 256 x 9
    qkv_gemm_kernel<<<g1, 128, 0, stream>>>(X16, WtA, b_attn, Qd, Kd, Vtd);

    flash_attn_kernel<<<BB * NH * (TSEQ / 16), 32, 0, stream>>>(Qd, Kd, Vtd, AO);

    dim3 g2(MTOT / 32, EMB / 256);     // 256 x 3
    proj_gemm_kernel<<<g2, 128, 0, stream>>>(AO, WtP, b_proj, (float*)d_out);
}
